// RelationHead_75857712382006
// MI455X (gfx1250) — compile-verified, matrix operand order checked by reading, not run on hardware
//
#include <hip/hip_runtime.h>
#include <hip/hip_bf16.h>

typedef __attribute__((ext_vector_type(16))) _Float16 v16h;
typedef __attribute__((ext_vector_type(8)))  _Float16 v8h;
typedef __attribute__((ext_vector_type(8)))  float    v8f;

#define HD 256
#define NQ 100
#define NQP 112   // NQ padded to multiple of 16
#define NQB 128   // NQ padded to multiple of 32 (for N-dim of g1 / K-dim of g2)
#define BS 4
#define NU 8
#define HW 16384  // 128*128

// ---------------------------------------------------------------------------
// r_ins0[b,q,d] = mean over 9 layers of query[l,q,b,d]
__global__ void meanq_k(const float* __restrict__ query, float* __restrict__ r0) {
    int q = blockIdx.x, b = blockIdx.y, d = threadIdx.x;
    float s = 0.f;
#pragma unroll
    for (int l = 0; l < 9; ++l)
        s += query[(((long long)l * NQ + q) * BS + b) * HD + d];
    r0[((long long)b * NQ + q) * HD + d] = s * (1.f / 9.f);
}

// ---------------------------------------------------------------------------
// r_ins1[b,q,g] = sum_f r_ins0[b,q,f] * dsc_n2r_W[q,f,g].
// r1f: fp32, 112 rows/batch (residual for g3); r1h: f16, 128 rows/batch
// (B operand of g1).  Rows q>=100 are zeroed (exact WMMA padding).
__global__ void matvec_k(const float* __restrict__ r0, const float* __restrict__ W,
                         float* __restrict__ r1f, _Float16* __restrict__ r1h) {
    int q = blockIdx.x;          // 0..127
    int g = threadIdx.x;         // 0..255
    if (q >= NQ) {
        for (int b = 0; b < BS; ++b) {
            if (q < NQP) r1f[((long long)b * NQP + q) * HD + g] = 0.f;
            r1h[((long long)b * NQB + q) * HD + g] = (_Float16)0.f;
        }
        return;
    }
    __shared__ float x[BS][HD];
    for (int b = 0; b < BS; ++b)
        x[b][g] = r0[((long long)b * NQ + q) * HD + g];
    __syncthreads();
    const float* w = W + (long long)q * HD * HD + g;
    float acc0 = 0.f, acc1 = 0.f, acc2 = 0.f, acc3 = 0.f;
    for (int f = 0; f < HD; ++f) {
        float wv = w[(long long)f * HD];
        acc0 += x[0][f] * wv; acc1 += x[1][f] * wv;
        acc2 += x[2][f] * wv; acc3 += x[3][f] * wv;
    }
    float acc[4] = {acc0, acc1, acc2, acc3};
    for (int b = 0; b < BS; ++b) {
        r1f[((long long)b * NQP + q) * HD + g] = acc[b];
        r1h[((long long)b * NQB + q) * HD + g] = (_Float16)acc[b];
    }
}

// ---------------------------------------------------------------------------
// Transpose-convert: out_f16[c][r] = in_f32[r][c] ;  R,C multiples of 32.
__global__ void tcvt_k(const float* __restrict__ in, _Float16* __restrict__ out,
                       int R, int C, long long sInZ, long long sOutZ) {
    in  += (long long)blockIdx.z * sInZ;
    out += (long long)blockIdx.z * sOutZ;
    __shared__ float t[32][33];
    int c0 = blockIdx.x * 32, r0 = blockIdx.y * 32;
    int tx = threadIdx.x;
    for (int i = threadIdx.y; i < 32; i += 8)
        t[i][tx] = in[(long long)(r0 + i) * C + c0 + tx];
    __syncthreads();
    for (int i = threadIdx.y; i < 32; i += 8)
        out[(long long)(c0 + i) * R + r0 + tx] = (_Float16)t[tx][i];
}

// ---------------------------------------------------------------------------
// Generic single-wave WMMA GEMM: C[m,n] = sum_k A[m,k] * BT[n,k]
// Each wave computes a 16x32 output tile (two 16x16 accumulators sharing the
// A fragment).  N must be a multiple of 32, K a multiple of 32 (K >= 32).
// Software-pipelined: the last K-step is peeled so the steady-state body is
// 6 unconditional b128 loads + 2 WMMA (partial s_wait_loadcnt, no branches).
// z index decomposes as zu = z / zsplit, zb = z % zsplit.
union V16U { v16h v; v8h h[2]; };

__global__ __launch_bounds__(32) void gemm16_k(
    const _Float16* __restrict__ A,  long long sAu,  long long sAb,  int lda,
    const _Float16* __restrict__ BT, long long sBu,  long long sBb,  int ldb,
    const float* __restrict__ bias,
    const float* __restrict__ resid, long long sRu,  int ldr,
    float* __restrict__ C,  long long sCu,  int ldc, int mstore,
    _Float16* __restrict__ C16, long long sC16u, long long sC16b, int ldc16,
    int zsplit, int K, int relu)
{
    int z  = blockIdx.z;
    int zu = z / zsplit, zb = z % zsplit;
    A  += zu * sAu + zb * sAb;
    BT += zu * sBu + zb * sBb;
    int n0 = blockIdx.x * 32, m0 = blockIdx.y * 16;
    int lane = threadIdx.x;
    int r16  = lane & 15, hi = lane >> 4;
    const _Float16* ap  = A  + (long long)(m0 + r16) * lda + hi * 8;
    const _Float16* bp0 = BT + (long long)(n0 + r16) * ldb + hi * 8;
    const _Float16* bp1 = bp0 + (long long)16 * ldb;

    v8f acc0 = {}, acc1 = {};
    v8h a0  = *(const v8h*)(ap);
    v8h a1  = *(const v8h*)(ap + 16);
    v8h b00 = *(const v8h*)(bp0);
    v8h b01 = *(const v8h*)(bp0 + 16);
    v8h b10 = *(const v8h*)(bp1);
    v8h b11 = *(const v8h*)(bp1 + 16);

    for (int k = 32; k < K; k += 32) {
        v8h na0  = *(const v8h*)(ap  + k);
        v8h na1  = *(const v8h*)(ap  + k + 16);
        v8h nb00 = *(const v8h*)(bp0 + k);
        v8h nb01 = *(const v8h*)(bp0 + k + 16);
        v8h nb10 = *(const v8h*)(bp1 + k);
        v8h nb11 = *(const v8h*)(bp1 + k + 16);
        __builtin_prefetch(bp0 + k + 32, 0, 3);
        __builtin_prefetch(bp1 + k + 32, 0, 3);
        V16U a, b;
        a.h[0] = a0;  a.h[1] = a1;
        b.h[0] = b00; b.h[1] = b01;
        acc0 = __builtin_amdgcn_wmma_f32_16x16x32_f16(
            false, a.v, false, b.v, (short)0, acc0, false, false);
        b.h[0] = b10; b.h[1] = b11;
        acc1 = __builtin_amdgcn_wmma_f32_16x16x32_f16(
            false, a.v, false, b.v, (short)0, acc1, false, false);
        a0 = na0; a1 = na1; b00 = nb00; b01 = nb01; b10 = nb10; b11 = nb11;
    }
    {
        V16U a, b;
        a.h[0] = a0;  a.h[1] = a1;
        b.h[0] = b00; b.h[1] = b01;
        acc0 = __builtin_amdgcn_wmma_f32_16x16x32_f16(
            false, a.v, false, b.v, (short)0, acc0, false, false);
        b.h[0] = b10; b.h[1] = b11;
        acc1 = __builtin_amdgcn_wmma_f32_16x16x32_f16(
            false, a.v, false, b.v, (short)0, acc1, false, false);
    }

#pragma unroll
    for (int t = 0; t < 2; ++t) {
        v8f acc = t ? acc1 : acc0;
        int col = n0 + t * 16 + r16;
        float bv = bias ? bias[col] : 0.f;
#pragma unroll
        for (int rr = 0; rr < 8; ++rr) {
            int row = m0 + rr + 8 * hi;
            float v = acc[rr] + bv;
            if (resid) v += resid[zu * sRu + (long long)row * ldr + col];
            if (relu)  v = v > 0.f ? v : 0.f;
            if (C && row < mstore) C[zu * sCu + (long long)row * ldc + col] = v;
            if (C16) C16[zu * sC16u + zb * sC16b + (long long)row * ldc16 + col] = (_Float16)v;
        }
    }
}

// ---------------------------------------------------------------------------
// s1[z,i] = sum_g h[i,g]*a1[g] ; s2 likewise.  hT is [z][g=256][i ld=128] f16.
__global__ void sdot_k(const _Float16* __restrict__ hT, const float* __restrict__ miga,
                       float* __restrict__ s1, float* __restrict__ s2) {
    int z = blockIdx.x, i = threadIdx.x;        // block 128
    if (i >= NQP) return;
    int u = z >> 2;
    const _Float16* hp = hT + (long long)z * (HD * 128) + i;
    const float* a1 = miga + u * 512;
    const float* a2 = a1 + HD;
    float acc1 = 0.f, acc2 = 0.f;
    for (int g = 0; g < HD; ++g) {
        float hv = (float)hp[(long long)g * 128];
        acc1 += hv * a1[g]; acc2 += hv * a2[g];
    }
    s1[z * NQP + i] = acc1; s2[z * NQP + i] = acc2;
}

// ---------------------------------------------------------------------------
// att[z][i][j] = softmax_j( leakyrelu(s1_i + s2_j, 0.2) ), f16, ld 128,
// j>=100 -> 0 (zero K-padding for att@h WMMA).
__global__ void softmax_k(const float* __restrict__ s1, const float* __restrict__ s2,
                          _Float16* __restrict__ att) {
    int i = blockIdx.x, z = blockIdx.y, j = threadIdx.x;   // block 128
    __shared__ float red[128];
    float s1v = s1[z * NQP + i];
    bool valid = j < NQ;
    float e;
    if (valid) {
        e = s1v + s2[z * NQP + j];
        e = e > 0.f ? e : 0.2f * e;
    } else e = -1e30f;
    red[j] = e; __syncthreads();
    for (int s = 64; s > 0; s >>= 1) { if (j < s) red[j] = fmaxf(red[j], red[j + s]); __syncthreads(); }
    float mx = red[0]; __syncthreads();
    float ex = valid ? __expf(e - mx) : 0.f;
    red[j] = ex; __syncthreads();
    for (int s = 64; s > 0; s >>= 1) { if (j < s) red[j] += red[j + s]; __syncthreads(); }
    float sum = red[0];
    att[(long long)z * (NQP * 128) + i * 128 + j] = (_Float16)(ex / sum);
}

// ---------------------------------------------------------------------------
// scores[b,q] = r[b,q,:] . sal_W + sal_b
__global__ void scores_k(const float* __restrict__ r, const float* __restrict__ salW,
                         const float* __restrict__ salb, float* __restrict__ out) {
    int q = blockIdx.x, b = blockIdx.y, g = threadIdx.x;
    __shared__ float red[256];
    red[g] = r[((long long)b * NQP + q) * HD + g] * salW[g];
    __syncthreads();
    for (int s = 128; s > 0; s >>= 1) { if (g < s) red[g] += red[g + s]; __syncthreads(); }
    if (g == 0) out[b * NQ + q] = red[0] + salb[0];
}

// ---------------------------------------------------------------------------
// LayerNorm + class head; writes dec as f16 (for MLP WMMA) and outputs_class.
__global__ void ln_cls_k(const float* __restrict__ r, const float* __restrict__ lng,
                         const float* __restrict__ lnb, const float* __restrict__ clsW,
                         const float* __restrict__ clsb, _Float16* __restrict__ dec16,
                         float* __restrict__ outcls) {
    int q = blockIdx.x, b = blockIdx.y, g = threadIdx.x;
    long long base = ((long long)b * NQP + q) * HD;
    float v = r[base + g];
    __shared__ float red[256];
    red[g] = v; __syncthreads();
    for (int s = 128; s > 0; s >>= 1) { if (g < s) red[g] += red[g + s]; __syncthreads(); }
    float mu = red[0] * (1.f / 256.f); __syncthreads();
    float d = v - mu;
    red[g] = d * d; __syncthreads();
    for (int s = 128; s > 0; s >>= 1) { if (g < s) red[g] += red[g + s]; __syncthreads(); }
    float var = red[0] * (1.f / 256.f); __syncthreads();
    float dec = lng[g] * d * rsqrtf(var + 1e-5f) + lnb[g];
    dec16[base + g] = (_Float16)dec;
    if (q < NQ) {
        red[g] = dec * clsW[g * 2]; __syncthreads();
        for (int s = 128; s > 0; s >>= 1) { if (g < s) red[g] += red[g + s]; __syncthreads(); }
        float c0 = red[0]; __syncthreads();
        red[g] = dec * clsW[g * 2 + 1]; __syncthreads();
        for (int s = 128; s > 0; s >>= 1) { if (g < s) red[g] += red[g + s]; __syncthreads(); }
        if (g == 0) {
            int o = (b * NQ + q) * 2;
            outcls[o] = c0 + clsb[0]; outcls[o + 1] = red[0] + clsb[1];
        }
    }
}

// ---------------------------------------------------------------------------
extern "C" void kernel_launch(void* const* d_in, const int* in_sizes, int n_in,
                              void* d_out, int out_size, void* d_ws, size_t ws_size,
                              hipStream_t stream) {
    (void)in_sizes; (void)n_in; (void)out_size; (void)ws_size;
    const float* query   = (const float*)d_in[0];
    const float* feat    = (const float*)d_in[1];
    const float* dscn2rW = (const float*)d_in[8];
    const float* migW    = (const float*)d_in[10];
    const float* miga    = (const float*)d_in[11];
    const float* fcW     = (const float*)d_in[12];
    const float* fcb     = (const float*)d_in[13];
    const float* salW    = (const float*)d_in[14];
    const float* salb    = (const float*)d_in[15];
    const float* clsW    = (const float*)d_in[16];
    const float* clsb    = (const float*)d_in[17];
    const float* W1      = (const float*)d_in[18];
    const float* b1      = (const float*)d_in[19];
    const float* W2      = (const float*)d_in[20];
    const float* b2      = (const float*)d_in[21];
    const float* W3      = (const float*)d_in[22];
    const float* b3      = (const float*)d_in[23];
    const float* lng     = (const float*)d_in[24];
    const float* lnb     = (const float*)d_in[25];
    float* out = (float*)d_out;
    float* out_scores = out;            // [4,100]
    float* out_cls    = out + 400;      // [4,100,2]
    float* out_mask   = out + 1200;     // [4,100,128,128]

    char* wp = (char*)d_ws;
    auto alloc = [&](size_t bytes) -> void* {
        void* p = (void*)wp;
        wp += (bytes + 255) & ~(size_t)255;
        return p;
    };
    float*     r0    = (float*)    alloc((size_t)BS * NQ  * HD * 4);
    float*     r1f   = (float*)    alloc((size_t)BS * NQP * HD * 4);
    _Float16*  r1h   = (_Float16*) alloc((size_t)BS * NQB * HD * 2);
    _Float16*  migWT = (_Float16*) alloc((size_t)NU * HD * HD * 2);
    _Float16*  fcWT  = (_Float16*) alloc((size_t)HD * (NU * HD) * 2);
    _Float16*  w1T   = (_Float16*) alloc((size_t)HD * HD * 2);
    _Float16*  w2T   = (_Float16*) alloc((size_t)HD * HD * 2);
    _Float16*  w3T   = (_Float16*) alloc((size_t)HD * HD * 2);
    _Float16*  hT    = (_Float16*) alloc((size_t)32 * HD * 128 * 2);
    float*     s1    = (float*)    alloc((size_t)32 * NQP * 4);
    float*     s2    = (float*)    alloc((size_t)32 * NQP * 4);
    _Float16*  att   = (_Float16*) alloc((size_t)32 * NQP * 128 * 2);
    _Float16*  Z     = (_Float16*) alloc((size_t)BS * NQP * (NU * HD) * 2);
    float*     rbuf  = (float*)    alloc((size_t)BS * NQP * HD * 4);
    _Float16*  dec16 = (_Float16*) alloc((size_t)BS * NQP * HD * 2);
    _Float16*  m1    = (_Float16*) alloc((size_t)BS * NQP * HD * 2);
    _Float16*  m2    = (_Float16*) alloc((size_t)BS * NQP * HD * 2);
    _Float16*  me    = (_Float16*) alloc((size_t)BS * NQP * HD * 2);
    _Float16*  featT = (_Float16*) alloc((size_t)HW * HD * 2);   // per-batch reuse

    // 1. mean over 9 layers
    meanq_k<<<dim3(NQ, BS), HD, 0, stream>>>(query, r0);
    // 2. per-query matvec r_ins1 (padded fp32 + f16)
    matvec_k<<<NQB, HD, 0, stream>>>(r0, dscn2rW, r1f, r1h);
    // 3. weight transposes -> f16
    tcvt_k<<<dim3(8, 8, NU), dim3(32, 8), 0, stream>>>(migW, migWT, HD, HD,
                                                       (long long)HD * HD, (long long)HD * HD);
    tcvt_k<<<dim3(8, 64, 1), dim3(32, 8), 0, stream>>>(fcW, fcWT, NU * HD, HD, 0, 0);
    tcvt_k<<<dim3(8, 8, 1), dim3(32, 8), 0, stream>>>(W1, w1T, HD, HD, 0, 0);
    tcvt_k<<<dim3(8, 8, 1), dim3(32, 8), 0, stream>>>(W2, w2T, HD, HD, 0, 0);
    tcvt_k<<<dim3(8, 8, 1), dim3(32, 8), 0, stream>>>(W3, w3T, HD, HD, 0, 0);

    // 4. g1: hT[z=u*4+b][g][j] = sum_f migWT[u][g,f] * r1h[b][j,f]   (h = x@W, transposed)
    //    N = 128 (r1h zero-padded), so hT's K-padding columns are computed zeros.
    gemm16_k<<<dim3(NQB / 32, HD / 16, 32), 32, 0, stream>>>(
        migWT, (long long)HD * HD, 0, HD,
        r1h, 0, (long long)NQB * HD, HD,
        nullptr, nullptr, 0, 0,
        nullptr, 0, 0, 0,
        hT, (long long)4 * HD * 128, (long long)HD * 128, 128,
        4, HD, 0);
    // 5. attention scores + softmax
    sdot_k<<<32, 128, 0, stream>>>(hT, miga, s1, s2);
    softmax_k<<<dim3(NQP, 32), 128, 0, stream>>>(s1, s2, att);
    // 6. g2: Z[b][i][u*256+g] = sum_j att[z][i,j] * hT[z][g,j]   (att@h)
    gemm16_k<<<dim3(HD / 32, NQP / 16, 32), 32, 0, stream>>>(
        att, (long long)4 * NQP * 128, (long long)NQP * 128, 128,
        hT, (long long)4 * HD * 128, (long long)HD * 128, 128,
        nullptr, nullptr, 0, 0,
        nullptr, 0, 0, 0,
        Z, (long long)HD, (long long)NQP * NU * HD, NU * HD,
        4, 128, 0);
    // 7. g3: r = relu(r_ins1 + Z @ fcWT^T + fc_b)
    gemm16_k<<<dim3(HD / 32, NQP / 16, BS), 32, 0, stream>>>(
        Z, (long long)NQP * NU * HD, 0, NU * HD,
        fcWT, 0, 0, NU * HD,
        fcb, r1f, (long long)NQP * HD, HD,
        rbuf, (long long)NQP * HD, HD, NQP,
        nullptr, 0, 0, 0,
        1, NU * HD, 1);
    // 8. saliency scores
    scores_k<<<dim3(NQ, BS), HD, 0, stream>>>(rbuf, salW, salb, out_scores);
    // 9. LayerNorm + class head (+ dec in f16)
    ln_cls_k<<<dim3(NQP, BS), HD, 0, stream>>>(rbuf, lng, lnb, clsW, clsb, dec16, out_cls);
    // 10. MLP (3 layers)
    gemm16_k<<<dim3(HD / 32, NQP / 16, BS), 32, 0, stream>>>(
        dec16, (long long)NQP * HD, 0, HD, w1T, 0, 0, HD,
        b1, nullptr, 0, 0, nullptr, 0, 0, 0,
        m1, (long long)NQP * HD, 0, HD, 1, HD, 1);
    gemm16_k<<<dim3(HD / 32, NQP / 16, BS), 32, 0, stream>>>(
        m1, (long long)NQP * HD, 0, HD, w2T, 0, 0, HD,
        b2, nullptr, 0, 0, nullptr, 0, 0, 0,
        m2, (long long)NQP * HD, 0, HD, 1, HD, 1);
    gemm16_k<<<dim3(HD / 32, NQP / 16, BS), 32, 0, stream>>>(
        m2, (long long)NQP * HD, 0, HD, w3T, 0, 0, HD,
        b3, nullptr, 0, 0, nullptr, 0, 0, 0,
        me, (long long)NQP * HD, 0, HD, 1, HD, 0);
    // 11. mask einsum per batch: out[b][q][hw] = sum_c me[b][q,c] * feat[b][c,hw]
    for (int b = 0; b < BS; ++b) {
        tcvt_k<<<dim3(HW / 32, HD / 32, 1), dim3(32, 8), 0, stream>>>(
            feat + (long long)b * HD * HW, featT, HD, HW, 0, 0);
        gemm16_k<<<dim3(HW / 32, NQP / 16, 1), 32, 0, stream>>>(
            me + (long long)b * NQP * HD, 0, 0, HD,
            featT, 0, 0, HD,
            nullptr, nullptr, 0, 0,
            out_mask + (long long)b * NQ * HW, 0, HW, NQ,
            nullptr, 0, 0, 0,
            1, HD, 0);
    }
}
